// MultiCriteriaGNNModel_56195352101027
// MI455X (gfx1250) — matrix operand order repaired and checked by reading, not compiled, not run on hardware
//
#include <hip/hip_runtime.h>

typedef __attribute__((ext_vector_type(16))) __bf16 v16bf;
typedef __attribute__((ext_vector_type(8)))  float  v8f;

#define N_O   50000
#define N_OP  5000
#define E_OO  400000
#define E_AS  200000
#define HID   64
#define KP    160   // padded K for edge MLP (132 -> 160 = 5 * 32)

struct Frag32 { uint4 lo; uint4 hi; };   // 32 bytes == v16bf

// ---------- helpers ----------
__device__ __forceinline__ unsigned short f2bf(float f) {
  unsigned u = __float_as_uint(f);
  unsigned r = u + 0x7FFFu + ((u >> 16) & 1u);   // RNE truncate to bf16
  return (unsigned short)(r >> 16);
}
__device__ __forceinline__ unsigned pack2bf(float a, float b) {
  return (unsigned)f2bf(a) | ((unsigned)f2bf(b) << 16);
}
// load a v16bf whose elements live at [base..base+15] (ushort units)
__device__ __forceinline__ v16bf load_afrag(const unsigned short* base) {
  Frag32 f;
  f.lo = *(const uint4*)(base);
  f.hi = *(const uint4*)(base + 8);
  return __builtin_bit_cast(v16bf, f);
}
// order-preserving float<->uint encoding for atomicMax on floats
__device__ __forceinline__ unsigned enc_f(float x) {
  unsigned u = __float_as_uint(x);
  return (u & 0x80000000u) ? ~u : (u | 0x80000000u);
}
__device__ __forceinline__ float dec_f(unsigned e) {
  unsigned u = (e & 0x80000000u) ? (e & 0x7FFFFFFFu) : ~e;
  return __uint_as_float(u);
}

// ---------- fills ----------
__global__ void k_fill_f32(float* p, float v, int n) {
  int i = blockIdx.x * 256 + threadIdx.x;
  if (i < n) p[i] = v;
}
__global__ void k_fill_u32(unsigned* p, unsigned v, int n) {
  int i = blockIdx.x * 256 + threadIdx.x;
  if (i < n) p[i] = v;
}

// ---------- input linear (tiny K: 10 or 7) + relu ----------
__global__ void k_lin_in(const float* __restrict__ X, const float* __restrict__ W,
                         const float* __restrict__ Bv, float* __restrict__ Y,
                         int N, int K) {
  int i = blockIdx.x * 256 + threadIdx.x;
  if (i >= N * HID) return;
  int n = i >> 6, j = i & 63;
  float s = Bv[j];
  for (int k = 0; k < K; ++k) s += X[n * K + k] * W[k * HID + j];
  Y[i] = s > 0.f ? s : 0.f;
}

// ---------- [N,64] @ [64,64] + bias via WMMA bf16 ----------
__global__ __launch_bounds__(256) void k_gemm64_bias(
    const float* __restrict__ X, const float* __restrict__ W,
    const float* __restrict__ Bv, float* __restrict__ Y, int N) {
  __shared__ v16bf sWf[8 * 32];                                 // 8 frags, frag-major, 8 KB
  __shared__ __align__(16) unsigned short sA[8 * 16 * 64];      // per-wave A tiles bf16, 16 KB
  int tid = threadIdx.x;
  int wave = tid >> 5, lane = tid & 31;
  int n16 = lane & 15, kh = lane >> 4;

  // stage W pre-swizzled into fragment order: frag = kbi*4 + t, 32 lanes x 16 elems
  {
    unsigned short* sW = (unsigned short*)sWf;
    for (int idx = tid; idx < 64 * 64; idx += 256) {
      int f = idx >> 9, l = (idx >> 4) & 31, i = idx & 15;
      int kbi = f >> 2, t = f & 3;
      int nl = l & 15, kl = l >> 4;
      int v = i >> 1, p = i & 1;
      int k = kbi * 32 + kl * 16 + v * 2 + p;                   // contiguous k = kl*16 + i, same order as A frag
      sW[idx] = f2bf(W[k * 64 + t * 16 + nl]);
    }
  }

  int mbase = blockIdx.x * 128 + wave * 16;
  bool full = (mbase + 16 <= N);                                // wave-uniform full-tile flag
  unsigned short* myA = sA + wave * 16 * 64;
  {
    int row = mbase + n16;
    int c0 = kh * 32;
    unsigned* dp = (unsigned*)(myA + n16 * 64 + c0);
    if (full || row < N) {
      const float4* sp = (const float4*)(X + (size_t)row * 64 + c0);
#pragma unroll
      for (int q = 0; q < 8; ++q) {
        float4 f = sp[q];
        dp[2 * q]     = pack2bf(f.x, f.y);
        dp[2 * q + 1] = pack2bf(f.z, f.w);
      }
    } else {
#pragma unroll
      for (int q = 0; q < 16; ++q) dp[q] = 0u;
    }
  }
  __syncthreads();

  v8f zero = {};
  v8f acc[4];
#pragma unroll
  for (int t = 0; t < 4; ++t) acc[t] = zero;

#pragma unroll
  for (int kbi = 0; kbi < 2; ++kbi) {
    // A frag: elems 0-15 at k=kb+kh*16..
    v16bf a = load_afrag(myA + n16 * 64 + kbi * 32 + kh * 16);
#pragma unroll
    for (int t = 0; t < 4; ++t) {
      v16bf b = sWf[(kbi * 4 + t) * 32 + lane];
      acc[t] = __builtin_amdgcn_wmma_f32_16x16x32_bf16(
          false, a, false, b, (short)0, acc[t], false, false);
    }
  }

  float bv[4];
#pragma unroll
  for (int t = 0; t < 4; ++t) bv[t] = Bv[t * 16 + n16];

  // per-lane base: row block (mbase + kh*8), column n16; C/D layout: M = kh*8 + rr
  float* ybase = Y + (size_t)(mbase + kh * 8) * 64 + n16;
  if (full) {
#pragma unroll
    for (int rr = 0; rr < 8; ++rr)
#pragma unroll
      for (int t = 0; t < 4; ++t)
        ybase[rr * 64 + t * 16] = acc[t][rr] + bv[t];
  } else {
#pragma unroll
    for (int rr = 0; rr < 8; ++rr) {
      if (mbase + kh * 8 + rr < N) {
#pragma unroll
        for (int t = 0; t < 4; ++t)
          ybase[rr * 64 + t * 16] = acc[t][rr] + bv[t];
      }
    }
  }
}

// ---------- GATv2 edge pass 1: alpha + segment max ----------
__global__ void k_edge_alpha(const float* __restrict__ xl, const float* __restrict__ xr,
                             const int* __restrict__ ei, const float* __restrict__ ea,
                             const float* __restrict__ we, const float* __restrict__ att,
                             float* __restrict__ alpha, unsigned* __restrict__ amax, int E) {
  int idx = blockIdx.x * 256 + threadIdx.x;
  if (idx >= E * 4) return;
  int e = idx >> 2, h = idx & 3;
  int s = ei[e], d = ei[E + e];
  float eav = ea[e];
  float accv = 0.f;
#pragma unroll
  for (int c = 0; c < 16; ++c) {
    int j = h * 16 + c;
    float mv = xl[(size_t)s * 64 + j] + xr[(size_t)d * 64 + j] + eav * we[j];
    mv = mv < 0.f ? 0.2f * mv : mv;                   // leaky_relu(0.2)
    accv += mv * att[j];
  }
  alpha[idx] = accv;
  atomicMax(&amax[d * 4 + h], enc_f(accv));
}

// ---------- pass 2: exp(alpha - max) + segment sum ----------
__global__ void k_edge_exp(float* __restrict__ alpha, const int* __restrict__ ei,
                           const unsigned* __restrict__ amax, float* __restrict__ den, int E) {
  int idx = blockIdx.x * 256 + threadIdx.x;
  if (idx >= E * 4) return;
  int e = idx >> 2, h = idx & 3;
  int d = ei[E + e];
  float ex = expf(alpha[idx] - dec_f(amax[d * 4 + h]));
  alpha[idx] = ex;
  atomicAdd(&den[d * 4 + h], ex);
}

// ---------- pass 3: weighted scatter-add of xl[src] ----------
__global__ void k_edge_scatter(const float* __restrict__ ex, const int* __restrict__ ei,
                               const float* __restrict__ den, const float* __restrict__ xl,
                               float* __restrict__ outacc, int E) {
  int idx = blockIdx.x * 256 + threadIdx.x;
  if (idx >= E * 4) return;
  int e = idx >> 2, h = idx & 3;
  int s = ei[e], d = ei[E + e];
  float a = ex[idx] / (den[d * 4 + h] + 1e-16f);
#pragma unroll
  for (int c = 0; c < 16; ++c) {
    int j = h * 16 + c;
    atomicAdd(&outacc[(size_t)d * 64 + j], xl[(size_t)s * 64 + j] * a);
  }
}

// ---------- h = relu(acc + bias1 [+ bias2]) ----------
__global__ void k_finalize(const float* __restrict__ acc, const float* __restrict__ b1,
                           const float* __restrict__ b2, float* __restrict__ h, int N) {
  int i = blockIdx.x * 256 + threadIdx.x;
  if (i >= N * HID) return;
  float v = acc[i] + b1[i & 63];
  if (b2) v += b2[i & 63];
  h[i] = v > 0.f ? v : 0.f;
}

// ---------- activation head: per op-node MLP (67 -> 64 -> 1), sigmoid ----------
__global__ void k_act_mlp(const float* __restrict__ hp, const float* __restrict__ u,
                          const float* __restrict__ w1, const float* __restrict__ b1,
                          const float* __restrict__ w2, const float* __restrict__ b2,
                          float* __restrict__ out) {
  __shared__ float sh[64];
  int n = blockIdx.x, t = threadIdx.x;
  float s = b1[t];
  for (int k = 0; k < 64; ++k) s += hp[(size_t)n * 64 + k] * w1[k * 64 + t];
  for (int k = 0; k < 3; ++k) s += u[k] * w1[(64 + k) * 64 + t];
  s = s > 0.f ? s : 0.f;
  sh[t] = s * w2[t];
  __syncthreads();
  for (int off = 32; off > 0; off >>= 1) {
    if (t < off) sh[t] += sh[t + off];
    __syncthreads();
  }
  if (t == 0) out[n] = 1.f / (1.f + expf(-(sh[0] + b2[0])));
}

// ---------- edge heads: concat(132) -> 64 relu -> 1 sigmoid via WMMA ----------
__global__ __launch_bounds__(256) void k_edge_mlp(
    const float* __restrict__ srcF, const float* __restrict__ dstF,
    const float* __restrict__ u, const float* __restrict__ ea,
    const int* __restrict__ ei, const float* __restrict__ w1,
    const float* __restrict__ b1, const float* __restrict__ w2,
    const float* __restrict__ b2, float* __restrict__ out, int E) {
  __shared__ v16bf sWf[20 * 32];                               // 20 frags, frag-major, 20 KB
  __shared__ __align__(16) unsigned short sA[8 * 16 * KP];     // 40 KB
  int tid = threadIdx.x;
  int wave = tid >> 5, lane = tid & 31;
  int n16 = lane & 15, kh = lane >> 4;

  // stage W1 pre-swizzled into fragment order (rows >= 132 are zero)
  {
    unsigned short* sW = (unsigned short*)sWf;
    for (int idx = tid; idx < KP * 64; idx += 256) {
      int f = idx >> 9, l = (idx >> 4) & 31, i = idx & 15;
      int kbi = f >> 2, t = f & 3;
      int nl = l & 15, kl = l >> 4;
      int v = i >> 1, p = i & 1;
      int k = kbi * 32 + kl * 16 + v * 2 + p;
      sW[idx] = (k < 132) ? f2bf(w1[k * 64 + t * 16 + nl]) : (unsigned short)0;
    }
  }

  int ebase = blockIdx.x * 128 + wave * 16;
  unsigned short* myA = sA + wave * 16 * KP;
  {
    int e = ebase + n16;
    unsigned* dp = (unsigned*)(myA + n16 * KP);                // 80 uints per row
    if (e < E) {
      int s = ei[e], d = ei[E + e];
      const float4* sp = (const float4*)(srcF + (size_t)s * 64);
      const float4* dpp = (const float4*)(dstF + (size_t)d * 64);
      if (kh == 0) {
        // cols 0..63 <- srcF row (16 float4)
#pragma unroll
        for (int q = 0; q < 16; ++q) {
          float4 f = sp[q];
          dp[2 * q]     = pack2bf(f.x, f.y);
          dp[2 * q + 1] = pack2bf(f.z, f.w);
        }
        // cols 64..79 <- dstF[0..15] (4 float4)
#pragma unroll
        for (int q = 0; q < 4; ++q) {
          float4 f = dpp[q];
          dp[32 + 2 * q]     = pack2bf(f.x, f.y);
          dp[32 + 2 * q + 1] = pack2bf(f.z, f.w);
        }
      } else {
        // cols 80..127 <- dstF[16..63] (12 float4)
#pragma unroll
        for (int q = 0; q < 12; ++q) {
          float4 f = dpp[4 + q];
          dp[40 + 2 * q]     = pack2bf(f.x, f.y);
          dp[40 + 2 * q + 1] = pack2bf(f.z, f.w);
        }
        // cols 128..131: u0,u1,u2,ea ; cols 132..159: zero
        dp[64] = pack2bf(u[0], u[1]);
        dp[65] = pack2bf(u[2], ea[e]);
#pragma unroll
        for (int q = 66; q < 80; ++q) dp[q] = 0u;
      }
    } else {
      unsigned* zp = dp + kh * 40;
#pragma unroll
      for (int q = 0; q < 40; ++q) zp[q] = 0u;
    }
  }
  __syncthreads();

  v8f zero = {};
  v8f acc[4];
#pragma unroll
  for (int t = 0; t < 4; ++t) acc[t] = zero;

#pragma unroll
  for (int kbi = 0; kbi < 5; ++kbi) {
    v16bf a = load_afrag(myA + n16 * KP + kbi * 32 + kh * 16);
#pragma unroll
    for (int t = 0; t < 4; ++t) {
      v16bf b = sWf[(kbi * 4 + t) * 32 + lane];
      acc[t] = __builtin_amdgcn_wmma_f32_16x16x32_bf16(
          false, a, false, b, (short)0, acc[t], false, false);
    }
  }

  // hidden = relu(acc + b1); scalar = hidden . w2 via cross-lane reduction
  float rowsum[8];
#pragma unroll
  for (int rr = 0; rr < 8; ++rr) rowsum[rr] = 0.f;
#pragma unroll
  for (int t = 0; t < 4; ++t) {
    float b1v = b1[t * 16 + n16];
    float w2v = w2[t * 16 + n16];
#pragma unroll
    for (int rr = 0; rr < 8; ++rr) {
      float h = acc[t][rr] + b1v;
      h = h > 0.f ? h : 0.f;
      float v = h * w2v;
      v += __shfl_xor(v, 1);
      v += __shfl_xor(v, 2);
      v += __shfl_xor(v, 4);
      v += __shfl_xor(v, 8);      // masks <= 8: stays within 16-lane half
      rowsum[rr] += v;
    }
  }
  if (n16 == 0) {
    float bb = b2[0];
#pragma unroll
    for (int rr = 0; rr < 8; ++rr) {
      int e = ebase + kh * 8 + rr;
      if (e < E) out[e] = 1.f / (1.f + expf(-(rowsum[rr] + bb)));
    }
  }
}

// ---------- host orchestration ----------
extern "C" void kernel_launch(void* const* d_in, const int* in_sizes, int n_in,
                              void* d_out, int out_size, void* d_ws, size_t ws_size,
                              hipStream_t stream) {
  (void)in_sizes; (void)n_in; (void)out_size; (void)ws_size;
  const float* x_order = (const float*)d_in[0];
  const float* x_op    = (const float*)d_in[1];
  const float* u       = (const float*)d_in[2];
  const float* ea[3]   = { (const float*)d_in[3], (const float*)d_in[4], (const float*)d_in[5] };
  const int*   ei[3]   = { (const int*)d_in[6], (const int*)d_in[7], (const int*)d_in[8] };
  const float* order_lin_w = (const float*)d_in[9];
  const float* order_lin_b = (const float*)d_in[10];
  const float* op_lin_w    = (const float*)d_in[11];
  const float* op_lin_b    = (const float*)d_in[12];
  const float *wl[3], *bl[3], *wr[3], *br[3], *att[3], *we[3], *bias[3];
  for (int t = 0; t < 3; ++t) {
    int b = 13 + t * 7;
    wl[t]   = (const float*)d_in[b + 0];
    bl[t]   = (const float*)d_in[b + 1];
    wr[t]   = (const float*)d_in[b + 2];
    br[t]   = (const float*)d_in[b + 3];
    att[t]  = (const float*)d_in[b + 4];
    we[t]   = (const float*)d_in[b + 5];
    bias[t] = (const float*)d_in[b + 6];
  }
  const float* act_w1 = (const float*)d_in[34];
  const float* act_b1 = (const float*)d_in[35];
  const float* act_w2 = (const float*)d_in[36];
  const float* act_b2 = (const float*)d_in[37];
  const float* asg_w1 = (const float*)d_in[38];
  const float* asg_b1 = (const float*)d_in[39];
  const float* asg_w2 = (const float*)d_in[40];
  const float* asg_b2 = (const float*)d_in[41];
  const float* seq_w1 = (const float*)d_in[42];
  const float* seq_b1 = (const float*)d_in[43];
  const float* seq_w2 = (const float*)d_in[44];
  const float* seq_b2 = (const float*)d_in[45];
  float* out = (float*)d_out;

  float* ws = (float*)d_ws;
  size_t off = 0;
  float* ho     = ws + off; off += (size_t)N_O  * HID;
  float* hp     = ws + off; off += (size_t)N_OP * HID;
  float* ho_acc = ws + off; off += (size_t)N_O  * HID;
  float* hp_acc = ws + off; off += (size_t)N_OP * HID;
  float* xl     = ws + off; off += (size_t)N_O  * HID;
  float* xr     = ws + off; off += (size_t)N_O  * HID;
  float* alpha  = ws + off; off += (size_t)E_OO * 4;
  unsigned* amax = (unsigned*)(ws + off); off += (size_t)N_O * 4;
  float* den    = ws + off; off += (size_t)N_O * 4;

  auto cdiv = [](int a, int b) { return (a + b - 1) / b; };
  const unsigned ENC_NEG_INF = 0x007FFFFFu;  // enc_f(-inf)

  k_lin_in<<<cdiv(N_O * HID, 256), 256, 0, stream>>>(x_order, order_lin_w, order_lin_b, ho, N_O, 10);
  k_lin_in<<<cdiv(N_OP * HID, 256), 256, 0, stream>>>(x_op, op_lin_w, op_lin_b, hp, N_OP, 7);

  for (int l = 0; l < 3; ++l) {
    k_fill_f32<<<cdiv(N_O * HID, 256), 256, 0, stream>>>(ho_acc, 0.f, N_O * HID);
    k_fill_f32<<<cdiv(N_OP * HID, 256), 256, 0, stream>>>(hp_acc, 0.f, N_OP * HID);

    const float* srcH[3] = { ho, hp, ho };
    const float* dstH[3] = { ho, ho, hp };
    int Nsrc[3] = { N_O, N_OP, N_O };
    int Ndst[3] = { N_O, N_O, N_OP };
    int Ecnt[3] = { E_OO, E_AS, E_AS };
    float* accb[3] = { ho_acc, ho_acc, hp_acc };

    for (int t = 0; t < 3; ++t) {
      int E = Ecnt[t];
      k_gemm64_bias<<<cdiv(Nsrc[t], 128), 256, 0, stream>>>(srcH[t], wl[t] + l * 4096, bl[t] + l * 64, xl, Nsrc[t]);
      k_gemm64_bias<<<cdiv(Ndst[t], 128), 256, 0, stream>>>(dstH[t], wr[t] + l * 4096, br[t] + l * 64, xr, Ndst[t]);
      k_fill_u32<<<cdiv(Ndst[t] * 4, 256), 256, 0, stream>>>(amax, ENC_NEG_INF, Ndst[t] * 4);
      k_fill_f32<<<cdiv(Ndst[t] * 4, 256), 256, 0, stream>>>(den, 0.f, Ndst[t] * 4);
      k_edge_alpha<<<cdiv(E * 4, 256), 256, 0, stream>>>(xl, xr, ei[t], ea[t], we[t] + l * 64, att[t] + l * 64, alpha, amax, E);
      k_edge_exp<<<cdiv(E * 4, 256), 256, 0, stream>>>(alpha, ei[t], amax, den, E);
      k_edge_scatter<<<cdiv(E * 4, 256), 256, 0, stream>>>(alpha, ei[t], den, xl, accb[t], E);
    }
    k_finalize<<<cdiv(N_O * HID, 256), 256, 0, stream>>>(ho_acc, bias[0] + l * 64, bias[1] + l * 64, ho, N_O);
    k_finalize<<<cdiv(N_OP * HID, 256), 256, 0, stream>>>(hp_acc, bias[2] + l * 64, nullptr, hp, N_OP);
  }

  // heads: out = [activation(5000) | assignment(200000) | sequence(400000)]
  k_act_mlp<<<N_OP, 64, 0, stream>>>(hp, u, act_w1, act_b1, act_w2, act_b2, out);
  k_edge_mlp<<<cdiv(E_AS, 128), 256, 0, stream>>>(hp, ho, u, ea[1], ei[1], asg_w1, asg_b1, asg_w2, asg_b2, out + N_OP, E_AS);
  k_edge_mlp<<<cdiv(E_OO, 128), 256, 0, stream>>>(ho, ho, u, ea[0], ei[0], seq_w1, seq_b1, seq_w2, seq_b2, out + N_OP + E_AS, E_OO);
}
